// k1_GNN_sub_ppgn_7842610283377
// MI455X (gfx1250) — matrix operand order repaired and checked by reading, not compiled, not run on hardware
//
#include <hip/hip_runtime.h>

// ---------------------------------------------------------------------------
// MI455X (gfx1250) nested-GNN. Heavy contractions on v_wmma_f32_16x16x32_f16.
// Weights pre-converted to f16 in fragment-friendly layouts; conv inputs staged
// transposed in LDS; conv weights streamed to LDS via the Tensor Data Mover.
// ---------------------------------------------------------------------------

typedef __attribute__((ext_vector_type(16))) _Float16 v16h;
typedef __attribute__((ext_vector_type(8)))  _Float16 v8h;
typedef __attribute__((ext_vector_type(8)))  float    v8f;
typedef __attribute__((ext_vector_type(4)))  unsigned int v4u;
typedef __attribute__((ext_vector_type(8)))  int v8i;
typedef __attribute__((ext_vector_type(4)))  int v4i;

#define GB 128
#define GN 32
#define GK 8
#define GH 32
#define N_SUB 4096
#define N_EDGES 65536
#define N_OEDGES 16384
#define PPGN 128
#define TSTR 264   // LDS tile row stride (halves): 132 dwords -> 4-bank skew

__device__ __forceinline__ int laneid() { return (int)(threadIdx.x & 31); }

__device__ __forceinline__ v8f wmma_f16(v16h a, v16h b, v8f c) {
  return __builtin_amdgcn_wmma_f32_16x16x32_f16(false, a, false, b, (short)0, c,
                                                false, false);
}
__device__ __forceinline__ v16h cat8(v8h lo, v8h hi) {
  return __builtin_shufflevector(lo, hi, 0, 1, 2, 3, 4, 5, 6, 7, 8, 9, 10, 11,
                                 12, 13, 14, 15);
}

// A 16x32 f16 fragment from row-major [rows][lda] (lda mult of 8 halves,
// col0 mult of 32): element j <-> k = col0 + (j>=8?16:0) + (L/16)*8 + (j&7).
// Per lane: two 16-byte vector loads (global or LDS).
__device__ __forceinline__ v16h load_A_f16v(const _Float16* A, int lda,
                                            int row0, int col0) {
  int l = laneid(); int m = (l & 15) + row0; int hh = l >> 4;
  const _Float16* p = A + (size_t)m * lda + col0 + hh * 8;
  v8h lo = *(const v8h*)p;
  v8h hi = *(const v8h*)(p + 16);
  return cat8(lo, hi);
}
// B 32x16 f16 fragment from an n-major store T[n][k] (row stride ldt halves,
// mult of 8; k0 mult of 32): element j <-> k = k0 + (j>=8?16:0) + (L/16)*8 + (j&7), n = L%16+n0.
__device__ __forceinline__ v16h load_B_t16v(const _Float16* T, int ldt, int k0,
                                            int n0) {
  int l = laneid(); int n = (l & 15) + n0; int hh = l >> 4;
  const _Float16* p = T + (size_t)n * ldt + k0 + hh * 8;
  v8h lo = *(const v8h*)p;
  v8h hi = *(const v8h*)(p + 16);
  return cat8(lo, hi);
}
// D 16x16 f32: element r <-> row = r + (L/16)*8, col = L%16
__device__ __forceinline__ void store_tile_f16(_Float16* O, int ldo, v8f acc) {
  int l = laneid(); int n = l & 15, hh = l >> 4;
#pragma unroll
  for (int r = 0; r < 8; ++r)
    O[(size_t)(r + hh * 8) * ldo + n] = (_Float16)acc[r];
}

// ---------------------------------------------------------------------------
// TDM: stream f16 weight matrix [128 rows][kpad cols] into LDS with per-row
// padding of 4 dwords so the LDS row stride is kpad+8 halves (bank skew).
// Issued by one wave; caller must __syncthreads() afterwards.
__device__ __forceinline__ void tdm_load_w(const _Float16* Wh, int kpad,
                                           _Float16* lds) {
#if __has_builtin(__builtin_amdgcn_tensor_load_to_lds)
  unsigned long long ga = (unsigned long long)(uintptr_t)Wh;
  unsigned int lo32 = (unsigned int)(uintptr_t)lds;  // LDS byte offset
  int pi = (kpad == 32) ? 3 : 5;  // pad_interval: 2^(pi+1) dwords == kpad/2
  v4u g0;
  g0[0] = 1u;                                   // count=1 (valid user D#)
  g0[1] = lo32;                                 // lds_addr
  g0[2] = (unsigned int)ga;                     // global_addr[31:0]
  g0[3] = (unsigned int)((ga >> 32) & 0x1FFFFFFu) | 0x80000000u;  // addr|type=2
  v8i g1;
  g1[0] = (int)((1u << 16) | (1u << 20) | ((unsigned)pi << 22) | (3u << 25));
  // ^ data_size=2B, pad_enable, pad_interval=pi, pad_amount=4 dwords
  g1[1] = (int)(((unsigned)kpad & 0xFFFFu) << 16);  // tensor_dim0 lo16
  g1[2] = (int)(128u << 16);                        // tensor_dim1 lo16
  g1[3] = (int)(((unsigned)kpad & 0xFFFFu) << 16);  // tile_dim0
  g1[4] = 128;                                      // tile_dim1 (tile_dim2=0)
  g1[5] = kpad;                                     // tensor_dim0_stride lo32
  g1[6] = 0;
  g1[7] = 0;
  v4i gz = {0, 0, 0, 0};
  v8i gz8 = {0, 0, 0, 0, 0, 0, 0, 0};
  __builtin_amdgcn_tensor_load_to_lds(g0, g1, gz, gz, gz8, 0);
  __builtin_amdgcn_s_wait_tensorcnt(0);
#else
  int l = laneid();
  int ldl = kpad + 8;
  int rowv = kpad >> 3;  // v8h chunks per row
  for (int i = l; i < 128 * rowv; i += 32) {
    int r = i / rowv, c = i % rowv;
    ((v8h*)(lds + (size_t)r * ldl))[c] = ((const v8h*)(Wh + (size_t)r * kpad))[c];
  }
#endif
}

// ---------------------------------------------------------------------------
// Kernel 1: a1 = relu(edge_attr @ nn1.w + nn1.b), f16 [E,128]
__global__ void k_edge_mlp1(const float* __restrict__ ea,
                            const float* __restrict__ w1,
                            const float* __restrict__ b1,
                            _Float16* __restrict__ a1) {
  int t = blockIdx.x * blockDim.x + threadIdx.x;
  if (t >= N_EDGES * 128) return;
  int e = t >> 7, k = t & 127;
  float acc = b1[k];
  const float* er = ea + (size_t)e * 5;
#pragma unroll
  for (int c = 0; c < 5; ++c) acc += er[c] * w1[c * 128 + k];
  a1[t] = (_Float16)fmaxf(acc, 0.0f);
}

// Weight prep: f32 [rows][cin] -> f16 [rows][kpad] zero-padded
__global__ void k_wpad(const float* __restrict__ in, int rows, int cin,
                       int kpad, _Float16* __restrict__ out) {
  int t = blockIdx.x * blockDim.x + threadIdx.x;
  if (t >= rows * kpad) return;
  int r = t / kpad, c = t % kpad;
  out[t] = (_Float16)((c < cin) ? in[(size_t)r * cin + c] : 0.0f);
}
// Weight prep: nn2.w f32 [128][ld] -> f16 transposed [ld][128]
__global__ void k_w2t(const float* __restrict__ w2, int ld,
                      _Float16* __restrict__ out) {
  int t = blockIdx.x * blockDim.x + threadIdx.x;
  if (t >= 128 * ld) return;
  int c = t >> 7, r = t & 127;
  out[t] = (_Float16)w2[(size_t)r * ld + c];
}

// ---------------------------------------------------------------------------
// Kernel 2: fused NNConv per subgraph (16 edges, 8 nodes in LDS).
__global__ void k_nnconv(const float* __restrict__ hin, int fin,
                         const _Float16* __restrict__ a1,
                         const _Float16* __restrict__ w2t,  // [fin*32][128] f16
                         const float* __restrict__ b2,
                         const float* __restrict__ root,
                         const float* __restrict__ bias,
                         const int* __restrict__ ei,
                         float* __restrict__ hout) {
  __shared__ float hsub[GK][GH];
  __shared__ float msg[16][GH];
  __shared__ int srcl[16], dstl[16];
  int s = blockIdx.x, tid = threadIdx.x;
  int wv = tid >> 5, lane = tid & 31;

  for (int idx = tid; idx < GK * fin; idx += 64)
    hsub[idx / fin][idx % fin] =
        hin[(size_t)(s * GK + idx / fin) * fin + idx % fin];
  if (tid < 16) {
    srcl[tid] = ei[s * 16 + tid] - s * GK;
    dstl[tid] = ei[N_EDGES + s * 16 + tid] - s * GK;
  }
  for (int idx = tid; idx < 16 * GH; idx += 64) (&msg[0][0])[idx] = 0.0f;
  __syncthreads();

  v16h af[4];
#pragma unroll
  for (int kk = 0; kk < 4; ++kk)
    af[kk] = load_A_f16v(a1 + (size_t)s * 16 * 128, 128, 0, kk * 32);

  int n = lane & 15, hh = lane >> 4;
  for (int t = wv; t < fin * 2; t += 2) {  // uniform per wave
    v8f acc = {};
#pragma unroll
    for (int kk = 0; kk < 4; ++kk)
      acc = wmma_f16(af[kk], load_B_t16v(w2t, 128, kk * 32, t * 16), acc);
    int i = t >> 1, ob = (t & 1) * 16;
#pragma unroll
    for (int r = 0; r < 8; ++r) {
      int e = r + hh * 8;
      float wval = acc[r] + b2[i * GH + ob + n];
      msg[e][ob + n] += wval * hsub[srcl[e]][i];
    }
  }
  __syncthreads();

  for (int idx = tid; idx < GK * GH; idx += 64) {
    int u = idx >> 5, o = idx & 31;
    float acc = bias[o];
    for (int e = 0; e < 16; ++e)
      if (dstl[e] == u) acc += msg[e][o];
    for (int i = 0; i < fin; ++i) acc += hsub[u][i] * root[i * GH + o];
    hout[(size_t)(s * GK + u) * GH + o] =
        acc > 0.0f ? acc : (__expf(acc) - 1.0f);
  }
}

// ---------------------------------------------------------------------------
// Kernel 3: fc1 then mean-pool K=8 nodes -> dn [4096,16]
__global__ void k_fc1_pool(const float* __restrict__ h,
                           const float* __restrict__ w,
                           const float* __restrict__ b,
                           float* __restrict__ dn) {
  int t = blockIdx.x * blockDim.x + threadIdx.x;
  if (t >= N_SUB * 16) return;
  int s = t >> 4, o = t & 15;
  float acc = 0.0f;
  for (int u = 0; u < GK; ++u) {
    const float* hp = h + (size_t)(s * GK + u) * GH;
    float a = b[o];
    for (int i = 0; i < GH; ++i) a += hp[i] * w[i * 16 + o];
    acc += a;
  }
  dn[t] = acc * 0.125f;
}

// ---------------------------------------------------------------------------
// Kernel 4: build z[b,ch,u,v] (Z[b,c,u,v] = t[b,v,u,c] after transpose(0,3,2,1))
__global__ void k_build_z(const float* __restrict__ pos,
                          const float* __restrict__ dn,
                          const int* __restrict__ oei,
                          const float* __restrict__ oea,
                          float* __restrict__ z) {
  int t = blockIdx.x * blockDim.x + threadIdx.x;
  if (t >= GB * GN * GN) return;
  int b = t >> 10, u = (t >> 5) & 31, v = t & 31;
  size_t base = ((size_t)b << 17) + (size_t)(u * GN + v);
  float e0 = 0, e1 = 0, e2 = 0, e3 = 0, e4 = 0;
  int mb = b * 128;
  for (int m = 0; m < 128; ++m) {
    int r = oei[mb + m] & 31;
    int c = oei[N_OEDGES + mb + m] & 31;
    if (c == u && r == v) {
      const float* a = oea + (size_t)(mb + m) * 5;
      e0 += a[0]; e1 += a[1]; e2 += a[2]; e3 += a[3]; e4 += 1.0f;
    }
  }
  z[base] = e0; z[base + 1024] = e1; z[base + 2048] = e2;
  z[base + 3072] = e3; z[base + 4096] = e4;
  float d = 0.0f;
  if (u != v) {
    const float* pu = pos + (size_t)(b * GN + u) * 3;
    const float* pv = pos + (size_t)(b * GN + v) * 3;
    float dx = pu[0] - pv[0], dy = pu[1] - pv[1], dz = pu[2] - pv[2];
    d = sqrtf(dx * dx + dy * dy + dz * dz);
  }
  z[base + (size_t)5 * 1024] = d;
#pragma unroll
  for (int f = 0; f < 16; ++f)
    z[base + (size_t)(6 + f) * 1024] =
        (u == v) ? dn[(size_t)(b * GN + u) * 16 + f] : 0.0f;
}

// ---------------------------------------------------------------------------
// Kernel 5: 1x1 conv + relu. Block = 256 thr (8 waves) = one 16-wide p-tile x
// all 128 cout. Input tile staged transposed in LDS; weights TDM'd into LDS.
__global__ void k_conv1x1(const float* __restrict__ inf,
                          const _Float16* __restrict__ inh, int cin, int kpad,
                          const _Float16* __restrict__ Wh,  // [128][kpad] f16
                          const float* __restrict__ bias,
                          _Float16* __restrict__ outh) {
  __shared__ _Float16 Wlds[128 * 136];     // rows stride kpad+8 (<=136)
  __shared__ _Float16 tile[16][TSTR];
  int pt = blockIdx.x;   // 0..63
  int b = blockIdx.y;
  int tid = threadIdx.x;
  size_t ibase = (size_t)b * PPGN * 1024 + (size_t)pt * 16;
  for (int idx = tid; idx < kpad * 16; idx += 256) {
    int ci = idx >> 4, n = idx & 15;
    float v = 0.0f;
    if (ci < cin)
      v = inf ? inf[ibase + (size_t)ci * 1024 + n]
              : (float)inh[ibase + (size_t)ci * 1024 + n];
    tile[n][ci] = (_Float16)v;
  }
  if ((tid >> 5) == 0) tdm_load_w(Wh, kpad, Wlds);
  __syncthreads();

  int wv = tid >> 5;  // co-tile 0..7
  v8f acc = {};
  int kt = kpad >> 5;
  for (int kk = 0; kk < kt; ++kk) {
    v16h a = load_A_f16v(Wlds, kpad + 8, wv * 16, kk * 32);
    v16h bb = load_B_t16v(&tile[0][0], TSTR, kk * 32, 0);
    acc = wmma_f16(a, bb, acc);
  }
  int lane = tid & 31, n = lane & 15, hh = lane >> 4;
  size_t obase = (size_t)b * PPGN * 1024 + (size_t)pt * 16 + n;
#pragma unroll
  for (int r = 0; r < 8; ++r) {
    int co = wv * 16 + hh * 8 + r;
    float v = acc[r] + bias[co];
    outh[obase + (size_t)co * 1024] = (_Float16)fmaxf(v, 0.0f);
  }
}

// ---------------------------------------------------------------------------
// Kernel 6: batched 32x32x32 matmul; m2 transposed through LDS per wave.
__global__ void k_mult(const _Float16* __restrict__ m1,
                       const _Float16* __restrict__ m2,
                       _Float16* __restrict__ mo) {
  __shared__ _Float16 mt[4][32 * 40];
  int wv = threadIdx.x >> 5, lane = threadIdx.x & 31;
  int bc = blockIdx.x * 4 + wv;
  const _Float16* A = m1 + (size_t)bc * 1024;
  const _Float16* B = m2 + (size_t)bc * 1024;
  _Float16* O = mo + (size_t)bc * 1024;
  _Float16* T = &mt[wv][0];
  const v8h* src = (const v8h*)B;
#pragma unroll
  for (int q = 0; q < 4; ++q) {
    v8h r = src[lane * 4 + q];  // row = lane, cols q*8..q*8+7
#pragma unroll
    for (int j = 0; j < 8; ++j) T[(q * 8 + j) * 40 + lane] = r[j];
  }
  __syncthreads();
  v16h a0 = load_A_f16v(A, 32, 0, 0);
  v16h a1 = load_A_f16v(A, 32, 16, 0);
  v16h b0 = load_B_t16v(T, 40, 0, 0);
  v16h b1 = load_B_t16v(T, 40, 0, 16);
  v8f zz = {};
  store_tile_f16(O, 32, wmma_f16(a0, b0, zz));
  store_tile_f16(O + 16, 32, wmma_f16(a0, b1, zz));
  store_tile_f16(O + 16 * 32, 32, wmma_f16(a1, b0, zz));
  store_tile_f16(O + 16 * 32 + 16, 32, wmma_f16(a1, b1, zz));
}

// ---------------------------------------------------------------------------
// Kernel 7: skip conv over concat([z f32 (cin1 ch), mult f16 (128 ch)]).
__global__ void k_skip(const float* __restrict__ zin, int cin1,
                       const _Float16* __restrict__ multh, int kpad,
                       const _Float16* __restrict__ Wh,  // [128][kpad] f16
                       const float* __restrict__ bias,
                       float* __restrict__ zout) {
  __shared__ _Float16 tile[16][TSTR];
  int pt = blockIdx.x;
  int b = blockIdx.y;
  int tid = threadIdx.x;
  int cint = cin1 + PPGN;
  size_t ibase = (size_t)b * PPGN * 1024 + (size_t)pt * 16;
  for (int idx = tid; idx < kpad * 16; idx += 256) {
    int k = idx >> 4, n = idx & 15;
    float v = 0.0f;
    if (k < cin1) v = zin[ibase + (size_t)k * 1024 + n];
    else if (k < cint) v = (float)multh[ibase + (size_t)(k - cin1) * 1024 + n];
    tile[n][k] = (_Float16)v;
  }
  __syncthreads();
  int wv = tid >> 5;
  v8f acc = {};
  int kt = kpad >> 5;
  for (int kk = 0; kk < kt; ++kk) {
    v16h a = load_A_f16v(Wh, kpad, wv * 16, kk * 32);
    v16h bb = load_B_t16v(&tile[0][0], TSTR, kk * 32, 0);
    acc = wmma_f16(a, bb, acc);
  }
  int lane = tid & 31, n = lane & 15, hh = lane >> 4;
  size_t obase = (size_t)b * PPGN * 1024 + (size_t)pt * 16 + n;
#pragma unroll
  for (int r = 0; r < 8; ++r) {
    int co = wv * 16 + hh * 8 + r;
    zout[obase + (size_t)co * 1024] = acc[r] + bias[co];
  }
}

// ---------------------------------------------------------------------------
__global__ void k_pool1(const float* __restrict__ z, float* __restrict__ pd,
                        float* __restrict__ po, float* __restrict__ pn) {
  int bc = blockIdx.x * blockDim.x + threadIdx.x;
  if (bc >= GB * PPGN) return;
  const float* m = z + (size_t)bc * 1024;
  float dmax = -1e30f, off = -1e30f, neg = -1e30f;
  for (int u = 0; u < GN; ++u)
    for (int v = 0; v < GN; ++v) {
      float x = m[u * GN + v];
      neg = fmaxf(neg, -x);
      if (u == v) dmax = fmaxf(dmax, x);
      else off = fmaxf(off, x);
    }
  pd[bc] = dmax; po[bc] = off; pn[bc] = neg;
}

__global__ void k_reduce_val(const float* __restrict__ pd,
                             const float* __restrict__ pn,
                             float* __restrict__ valp) {
  __shared__ float sd[256], sn[256];
  int t = threadIdx.x;
  float md = -1e30f, mn = -1e30f;
  for (int i = t; i < GB * PPGN; i += 256) {
    md = fmaxf(md, pd[i]); mn = fmaxf(mn, pn[i]);
  }
  sd[t] = md; sn[t] = mn;
  __syncthreads();
  for (int s = 128; s > 0; s >>= 1) {
    if (t < s) { sd[t] = fmaxf(sd[t], sd[t + s]); sn[t] = fmaxf(sn[t], sn[t + s]); }
    __syncthreads();
  }
  if (t == 0) valp[0] = fabsf(sd[0] + sn[0]);
}

__global__ void k_final(const float* __restrict__ pd, const float* __restrict__ po,
                        const float* __restrict__ valp,
                        const float* __restrict__ g1w, const float* __restrict__ g1b,
                        const float* __restrict__ g2w, const float* __restrict__ g2b,
                        float* __restrict__ out) {
  __shared__ float zp[2 * PPGN];
  __shared__ float hid[PPGN];
  int b = blockIdx.x, t = threadIdx.x;
  float val = valp[0];
  float d = pd[b * PPGN + t], o = po[b * PPGN + t];
  zp[t] = d;
  zp[PPGN + t] = fmaxf(o, d - val);
  __syncthreads();
  float a = g1b[t];
  for (int c = 0; c < 2 * PPGN; ++c) a += zp[c] * g1w[c * PPGN + t];
  hid[t] = fmaxf(a, 0.0f) * g2w[t];
  __syncthreads();
  for (int s = PPGN / 2; s > 0; s >>= 1) {
    if (t < s) hid[t] += hid[t + s];
    __syncthreads();
  }
  if (t == 0) out[b] = hid[0] + g2b[0];
}

// ---------------------------------------------------------------------------
enum {
  IN_X = 0, IN_EI, IN_EA, IN_N2S, IN_S2G, IN_OEI, IN_OEA, IN_OPOS,
  B0_M1C0_B = 8, B0_M1C0_W, B0_M1C1_B, B0_M1C1_W,
  B0_M2C0_B, B0_M2C0_W, B0_M2C1_B, B0_M2C1_W, B0_SKIP_B, B0_SKIP_W,
  B1_M1C0_B, B1_M1C0_W, B1_M1C1_B, B1_M1C1_W,
  B1_M2C0_B, B1_M2C0_W, B1_M2C1_B, B1_M2C1_W, B1_SKIP_B, B1_SKIP_W,
  C0_BIAS, C0_NN1B, C0_NN1W, C0_NN2B, C0_NN2W, C0_ROOT,
  C1_BIAS, C1_NN1B, C1_NN1W, C1_NN2B, C1_NN2W, C1_ROOT,
  C2_BIAS, C2_NN1B, C2_NN1W, C2_NN2B, C2_NN2W, C2_ROOT,
  FC1_B, FC1_W, GFC1_B, GFC1_W, GFC2_B, GFC2_W
};

extern "C" void kernel_launch(void* const* d_in, const int* in_sizes, int n_in,
                              void* d_out, int out_size, void* d_ws,
                              size_t ws_size, hipStream_t stream) {
  (void)in_sizes; (void)n_in; (void)out_size; (void)ws_size;
  const float* x    = (const float*)d_in[IN_X];
  const int*   ei   = (const int*)d_in[IN_EI];
  const float* ea   = (const float*)d_in[IN_EA];
  const int*   oei  = (const int*)d_in[IN_OEI];
  const float* oea  = (const float*)d_in[IN_OEA];
  const float* opos = (const float*)d_in[IN_OPOS];
  auto F = [&](int i) { return (const float*)d_in[i]; };

  char* p = (char*)d_ws;
  auto take = [&](size_t bytes) {
    char* r = p; p += (bytes + 255) & ~(size_t)255; return r;
  };
  float*    dn   = (float*)take((size_t)N_SUB * 16 * 4);
  float*    zA   = (float*)take((size_t)GB * PPGN * 1024 * 4);
  char*     zBr  =         take((size_t)GB * PPGN * 1024 * 4);
  float*    zB   = (float*)zBr;
  _Float16* m1   = (_Float16*)take((size_t)GB * PPGN * 1024 * 2);
  _Float16* m2   = (_Float16*)take((size_t)GB * PPGN * 1024 * 2);
  _Float16* mult = (_Float16*)take((size_t)GB * PPGN * 1024 * 2);
  float*    pd   = (float*)take((size_t)GB * PPGN * 4);
  float*    po   = (float*)take((size_t)GB * PPGN * 4);
  float*    pn   = (float*)take((size_t)GB * PPGN * 4);
  float*    valp = (float*)take(256);
  _Float16* w2t  = (_Float16*)take((size_t)1024 * 128 * 2);
  // padded f16 conv weights
  _Float16* wq_b0c0  = (_Float16*)take((size_t)128 * 32 * 2);
  _Float16* wq_b0c1  = (_Float16*)take((size_t)128 * 128 * 2);
  _Float16* wq_b0d0  = (_Float16*)take((size_t)128 * 32 * 2);
  _Float16* wq_b0d1  = (_Float16*)take((size_t)128 * 128 * 2);
  _Float16* wq_b0s   = (_Float16*)take((size_t)128 * 160 * 2);
  _Float16* wq_b1c0  = (_Float16*)take((size_t)128 * 128 * 2);
  _Float16* wq_b1c1  = (_Float16*)take((size_t)128 * 128 * 2);
  _Float16* wq_b1d0  = (_Float16*)take((size_t)128 * 128 * 2);
  _Float16* wq_b1d1  = (_Float16*)take((size_t)128 * 128 * 2);
  _Float16* wq_b1s   = (_Float16*)take((size_t)128 * 256 * 2);
  // Stage-A scratch aliases zB (first written at skip of block 1).
  _Float16* a1 = (_Float16*)zBr;                           // 16 MiB
  float*    hA = (float*)(zBr + (size_t)20 * 1024 * 1024); // 4 MiB
  float*    hB = (float*)(zBr + (size_t)26 * 1024 * 1024); // 4 MiB

  auto wpad = [&](int idx, int cin, int kpad, _Float16* dst) {
    k_wpad<<<(128 * kpad + 255) / 256, 256, 0, stream>>>(F(idx), 128, cin,
                                                         kpad, dst);
  };
  // ---- weight prep (f16, padded / transposed) ----
  wpad(B0_M1C0_W, 22, 32, wq_b0c0);  wpad(B0_M1C1_W, 128, 128, wq_b0c1);
  wpad(B0_M2C0_W, 22, 32, wq_b0d0);  wpad(B0_M2C1_W, 128, 128, wq_b0d1);
  wpad(B0_SKIP_W, 150, 160, wq_b0s);
  wpad(B1_M1C0_W, 128, 128, wq_b1c0); wpad(B1_M1C1_W, 128, 128, wq_b1c1);
  wpad(B1_M2C0_W, 128, 128, wq_b1d0); wpad(B1_M2C1_W, 128, 128, wq_b1d1);
  wpad(B1_SKIP_W, 256, 256, wq_b1s);

  // ---- NNConv stack ----
  k_edge_mlp1<<<(N_EDGES * 128) / 256, 256, 0, stream>>>(ea, F(C0_NN1W), F(C0_NN1B), a1);
  k_w2t<<<(128 * 480 + 255) / 256, 256, 0, stream>>>(F(C0_NN2W), 480, w2t);
  k_nnconv<<<N_SUB, 64, 0, stream>>>(x, 15, a1, w2t, F(C0_NN2B), F(C0_ROOT),
                                     F(C0_BIAS), ei, hA);
  k_edge_mlp1<<<(N_EDGES * 128) / 256, 256, 0, stream>>>(ea, F(C1_NN1W), F(C1_NN1B), a1);
  k_w2t<<<(128 * 1024 + 255) / 256, 256, 0, stream>>>(F(C1_NN2W), 1024, w2t);
  k_nnconv<<<N_SUB, 64, 0, stream>>>(hA, 32, a1, w2t, F(C1_NN2B), F(C1_ROOT),
                                     F(C1_BIAS), ei, hB);
  k_edge_mlp1<<<(N_EDGES * 128) / 256, 256, 0, stream>>>(ea, F(C2_NN1W), F(C2_NN1B), a1);
  k_w2t<<<(128 * 1024 + 255) / 256, 256, 0, stream>>>(F(C2_NN2W), 1024, w2t);
  k_nnconv<<<N_SUB, 64, 0, stream>>>(hB, 32, a1, w2t, F(C2_NN2B), F(C2_ROOT),
                                     F(C2_BIAS), ei, hA);
  k_fc1_pool<<<(N_SUB * 16) / 256, 256, 0, stream>>>(hA, F(FC1_W), F(FC1_B), dn);

  // ---- dense z ----
  k_build_z<<<(GB * GN * GN) / 256, 256, 0, stream>>>(opos, dn, oei, oea, zA);

  dim3 cg(64, GB);  // (p-tiles, graphs)
  // ---- PPGN block 1 ----
  k_conv1x1<<<cg, 256, 0, stream>>>(zA, (const _Float16*)nullptr, 22, 32,
                                    wq_b0c0, F(B0_M1C0_B), mult);
  k_conv1x1<<<cg, 256, 0, stream>>>((const float*)nullptr, mult, 128, 128,
                                    wq_b0c1, F(B0_M1C1_B), m1);
  k_conv1x1<<<cg, 256, 0, stream>>>(zA, (const _Float16*)nullptr, 22, 32,
                                    wq_b0d0, F(B0_M2C0_B), mult);
  k_conv1x1<<<cg, 256, 0, stream>>>((const float*)nullptr, mult, 128, 128,
                                    wq_b0d1, F(B0_M2C1_B), m2);
  k_mult<<<(GB * PPGN) / 4, 128, 0, stream>>>(m1, m2, mult);
  k_skip<<<cg, 256, 0, stream>>>(zA, 22, mult, 160, wq_b0s, F(B0_SKIP_B), zB);

  // ---- PPGN block 2 ----
  k_conv1x1<<<cg, 256, 0, stream>>>(zB, (const _Float16*)nullptr, 128, 128,
                                    wq_b1c0, F(B1_M1C0_B), mult);
  k_conv1x1<<<cg, 256, 0, stream>>>((const float*)nullptr, mult, 128, 128,
                                    wq_b1c1, F(B1_M1C1_B), m1);
  k_conv1x1<<<cg, 256, 0, stream>>>(zB, (const _Float16*)nullptr, 128, 128,
                                    wq_b1d0, F(B1_M2C0_B), mult);
  k_conv1x1<<<cg, 256, 0, stream>>>((const float*)nullptr, mult, 128, 128,
                                    wq_b1d1, F(B1_M2C1_B), m2);
  k_mult<<<(GB * PPGN) / 4, 128, 0, stream>>>(m1, m2, mult);
  k_skip<<<cg, 256, 0, stream>>>(zB, 128, mult, 256, wq_b1s, F(B1_SKIP_B), zA);

  // ---- pooling + readout ----
  k_pool1<<<(GB * PPGN) / 256, 256, 0, stream>>>(zA, pd, po, pn);
  k_reduce_val<<<1, 256, 0, stream>>>(pd, pn, valp);
  k_final<<<GB, PPGN, 0, stream>>>(pd, po, valp, F(GFC1_W), F(GFC1_B),
                                   F(GFC2_W), F(GFC2_B), (float*)d_out);
}
